// RnnKF_49830210568554
// MI455X (gfx1250) — matrix operand order repaired and checked by reading, not compiled, not run on hardware
//
#include <hip/hip_runtime.h>
#include <hip/hip_bf16.h>

// RnnKF on MI455X (gfx1250, wave32, WMMA).
//
// Reformulation: with G = I - H^T K^T (so G^T = I - K H):
//   s_t = s_{t-1} @ (F^T G) + (Bu_t @ G + Ko_t)      <- only sequential part
//   y_t = (s_{t-1} @ F^T + Bu_t) @ H^T               <- parallel, post-hoc
// Phases:
//   P0: Gt = I - K@H ; Mt = Gt@F            (tiny 256x256, scalar)
//   P1: Bu = inputs @ B^T                   (WMMA GEMM, M=65536)
//   P2: Ko = obs @ K^T ; c = Bu@Gt^T + Ko   (WMMA GEMMs)
//   P3: s_t = s_{t-1}@Mt^T + c_t            (persistent 1-WG WMMA recurrence,
//                                            Mt strip in VGPRs, state in LDS)
//   P4: tmp = sprev@F^T + Bu ; y = tmp@H^T  (WMMA GEMMs)
//
// Workspace layout (floats): Bu | Cbuf | Sprev | Gt | Mt  (~193 MB).

#define S_DIM   256
#define BATCH_N 64
#define T_STEPS 1024
#define M_ROWS  (BATCH_N * T_STEPS)   // 65536

typedef __attribute__((ext_vector_type(16))) __bf16 v16bf;
typedef __attribute__((ext_vector_type(8)))  __bf16 v8bf;
typedef __attribute__((ext_vector_type(8)))  float  v8f;

// ---------- WMMA tile loaders (wave32 layouts per CDNA5 ISA 7.12.2) ----------

// A-matrix 16x32 bf16: lane<16 -> M=lane, K = k0+[0..7] and k0+[16..23];
//                      lane>=16 -> M=lane-16, K = k0+[8..15] and k0+[24..31].
static __device__ __forceinline__ v16bf load_a_f32(const float* __restrict__ A,
                                                   int lda, int row0, int k0) {
  const int lane = threadIdx.x & 31;
  const int r  = row0 + (lane & 15);
  const int kb = k0 + ((lane >> 4) << 3);       // 0 or 8
  const float* p = A + (size_t)r * lda + kb;
  float4 f0 = *(const float4*)(p);
  float4 f1 = *(const float4*)(p + 4);
  float4 f2 = *(const float4*)(p + 16);
  float4 f3 = *(const float4*)(p + 20);
  float t[16] = { f0.x, f0.y, f0.z, f0.w, f1.x, f1.y, f1.z, f1.w,
                  f2.x, f2.y, f2.z, f2.w, f3.x, f3.y, f3.z, f3.w };
  v16bf a;
#pragma unroll
  for (int i = 0; i < 16; ++i) a[i] = (__bf16)t[i];
  return a;
}

// A-matrix from bf16 LDS (same layout), 16B vector loads.
static __device__ __forceinline__ v16bf load_a_lds(const __bf16* __restrict__ S,
                                                   int srow, int row0, int k0) {
  const int lane = threadIdx.x & 31;
  const int r  = row0 + (lane & 15);
  const int kb = k0 + ((lane >> 4) << 3);
  const __bf16* p = S + r * srow + kb;
  v8bf lo = *(const v8bf*)(p);
  v8bf hi = *(const v8bf*)(p + 16);
  return __builtin_shufflevector(lo, hi, 0,1,2,3,4,5,6,7,8,9,10,11,12,13,14,15);
}

// B-matrix 32x16 bf16 from W row-major [N,K] (computing X @ W^T):
// lane<16 -> col=lane, K=k0+[0..7],[16..23]; lane>=16 -> col=lane-16, K=k0+[8..15],[24..31].
static __device__ __forceinline__ v16bf load_b_f32(const float* __restrict__ W,
                                                   int ldw, int n0, int k0) {
  const int lane = threadIdx.x & 31;
  const int c  = n0 + (lane & 15);
  const int kb = k0 + ((lane >> 4) << 3);       // 0 or 8
  const float* p = W + (size_t)c * ldw + kb;
  float4 f0 = *(const float4*)(p);
  float4 f1 = *(const float4*)(p + 4);
  float4 f2 = *(const float4*)(p + 16);
  float4 f3 = *(const float4*)(p + 20);
  float t[16] = { f0.x, f0.y, f0.z, f0.w, f1.x, f1.y, f1.z, f1.w,
                  f2.x, f2.y, f2.z, f2.w, f3.x, f3.y, f3.z, f3.w };
  v16bf b;
#pragma unroll
  for (int i = 0; i < 16; ++i) b[i] = (__bf16)t[i];
  return b;
}

// C/D 16x16 f32: VGPR i, lane<16 -> (M=i, N=lane); lane>=16 -> (M=8+i, N=lane-16).
static __device__ __forceinline__ void add_c(const float* __restrict__ C, int ldc,
                                             int row0, int col0, v8f& acc) {
  const int lane = threadIdx.x & 31;
  const int c  = col0 + (lane & 15);
  const int rb = row0 + ((lane >> 4) << 3);
#pragma unroll
  for (int i = 0; i < 8; ++i) acc[i] += C[(size_t)(rb + i) * ldc + c];
}

static __device__ __forceinline__ void store_c(float* __restrict__ O, int ldo,
                                               int row0, int col0, v8f acc) {
  const int lane = threadIdx.x & 31;
  const int c  = col0 + (lane & 15);
  const int rb = row0 + ((lane >> 4) << 3);
#pragma unroll
  for (int i = 0; i < 8; ++i) O[(size_t)(rb + i) * ldo + c] = acc[i];
}

static __device__ __forceinline__ v8f wmma_bf16(v16bf a, v16bf b, v8f c) {
  return __builtin_amdgcn_wmma_f32_16x16x32_bf16(false, a, false, b,
                                                 (short)0, c, false, false);
}

// ---------- Parallel GEMM: Out[M,N] = A[M,K] @ W[N,K]^T (+ Cadd) ----------
// One wave per 32x32 output tile; 8 waves per block.
__global__ __launch_bounds__(256) void gemm_wt_bf16(
    const float* __restrict__ A, const float* __restrict__ W,
    const float* __restrict__ Cadd, float* __restrict__ Out,
    int M, int N, int K) {
  const int wave   = (blockIdx.x * blockDim.x + threadIdx.x) >> 5;
  const int tilesN = N >> 5;
  const int tn = (wave % tilesN) << 5;
  const int tm = (wave / tilesN) << 5;
  if (tm >= M) return;

  v8f a00 = {}, a01 = {}, a10 = {}, a11 = {};
#pragma unroll
  for (int k = 0; k < 256; k += 32) {
    v16bf A0 = load_a_f32(A, K, tm,      k);
    v16bf A1 = load_a_f32(A, K, tm + 16, k);
    v16bf B0 = load_b_f32(W, K, tn,      k);
    v16bf B1 = load_b_f32(W, K, tn + 16, k);
    a00 = wmma_bf16(A0, B0, a00);
    a01 = wmma_bf16(A0, B1, a01);
    a10 = wmma_bf16(A1, B0, a10);
    a11 = wmma_bf16(A1, B1, a11);
  }
  if (Cadd) {
    add_c(Cadd, N, tm,      tn,      a00);
    add_c(Cadd, N, tm,      tn + 16, a01);
    add_c(Cadd, N, tm + 16, tn,      a10);
    add_c(Cadd, N, tm + 16, tn + 16, a11);
  }
  store_c(Out, N, tm,      tn,      a00);
  store_c(Out, N, tm,      tn + 16, a01);
  store_c(Out, N, tm + 16, tn,      a10);
  store_c(Out, N, tm + 16, tn + 16, a11);
}

// ---------- Tiny 256x256 precomputes (negligible FLOPs) ----------
__global__ void gt_kernel(const float* __restrict__ K, const float* __restrict__ H,
                          float* __restrict__ Gt) {  // Gt = I - K@H
  const int n = blockIdx.x, k = threadIdx.x;
  float acc = 0.f;
  for (int o = 0; o < S_DIM; ++o) acc += K[n * S_DIM + o] * H[o * S_DIM + k];
  Gt[n * S_DIM + k] = (n == k ? 1.f : 0.f) - acc;
}

__global__ void mt_kernel(const float* __restrict__ Gt, const float* __restrict__ F,
                          float* __restrict__ Mt) {  // Mt = Gt@F
  const int n = blockIdx.x, k = threadIdx.x;
  float acc = 0.f;
  for (int o = 0; o < S_DIM; ++o) acc += Gt[n * S_DIM + o] * F[o * S_DIM + k];
  Mt[n * S_DIM + k] = acc;
}

__global__ void init_copy(const float* __restrict__ init, float* __restrict__ Sprev) {
  const int idx = blockIdx.x * blockDim.x + threadIdx.x;   // 64*256
  const int b = idx >> 8, s = idx & 255;
  Sprev[((size_t)b * T_STEPS) * S_DIM + s] = init[idx];    // sprev[b, t=0]
}

// ---------- Sequential recurrence: s_t = s_{t-1} @ Mt^T + c_t ----------
// One persistent workgroup, 16 waves; wave w owns output cols [16w,16w+16).
// Mt strip (256x16 bf16 = 8 B-tiles) pinned in VGPRs; state double-buffered
// in LDS as bf16 (row stride 264 to stagger banks).
#define SROW 264
__global__ __launch_bounds__(512, 1) void kf_recurrence(
    const float* __restrict__ init, const float* __restrict__ Cseq,
    const float* __restrict__ Mt, float* __restrict__ OutStates,
    float* __restrict__ Sprev) {
  __shared__ __bf16 sbuf[2 * BATCH_N * SROW];
  const int wave = threadIdx.x >> 5;
  const int lane = threadIdx.x & 31;
  const int n0 = wave << 4;

  v16bf bt[8];
#pragma unroll
  for (int kk = 0; kk < 8; ++kk) bt[kk] = load_b_f32(Mt, S_DIM, n0, kk * 32);

  for (int idx = threadIdx.x; idx < BATCH_N * S_DIM; idx += 512) {
    const int b = idx >> 8, s = idx & 255;
    sbuf[b * SROW + s] = (__bf16)init[idx];
  }
  __syncthreads();

  const int col = n0 + (lane & 15);
  const int rb8 = (lane >> 4) << 3;
  int cur = 0;
  for (int t = 0; t < T_STEPS; ++t) {
    const __bf16* sb = sbuf + cur * (BATCH_N * SROW);
    __bf16* sn = sbuf + (cur ^ 1) * (BATCH_N * SROW);
#pragma unroll
    for (int m = 0; m < 4; ++m) {
      v8f acc = {};
#pragma unroll
      for (int kk = 0; kk < 8; ++kk)
        acc = wmma_bf16(load_a_lds(sb, SROW, m * 16, kk * 32), bt[kk], acc);
#pragma unroll
      for (int i = 0; i < 8; ++i) {
        const int b = m * 16 + rb8 + i;
        const size_t off = ((size_t)b * T_STEPS + t) * S_DIM + col;
        const float v = acc[i] + Cseq[off];
        OutStates[off] = v;                        // state_estimation[b,t]
        if (t + 1 < T_STEPS) Sprev[off + S_DIM] = v;  // sprev[b,t+1]
        sn[b * SROW + col] = (__bf16)v;
      }
    }
    __syncthreads();
    cur ^= 1;
  }
}

// ---------- Launch ----------
extern "C" void kernel_launch(void* const* d_in, const int* in_sizes, int n_in,
                              void* d_out, int out_size, void* d_ws, size_t ws_size,
                              hipStream_t stream) {
  (void)in_sizes; (void)n_in; (void)out_size; (void)ws_size;
  const float* init = (const float*)d_in[0];
  const float* inp  = (const float*)d_in[1];
  const float* obs  = (const float*)d_in[2];
  const float* Fm   = (const float*)d_in[3];
  const float* Bm   = (const float*)d_in[4];
  const float* Hm   = (const float*)d_in[5];
  const float* Km   = (const float*)d_in[6];

  float* out_states = (float*)d_out;
  float* out_obs    = out_states + (size_t)M_ROWS * S_DIM;

  float* ws    = (float*)d_ws;
  float* Bu    = ws;
  float* Cbuf  = Bu   + (size_t)M_ROWS * S_DIM;
  float* Sprev = Cbuf + (size_t)M_ROWS * S_DIM;
  float* Gt    = Sprev + (size_t)M_ROWS * S_DIM;
  float* Mt    = Gt + S_DIM * S_DIM;

  const int tiles  = (M_ROWS / 32) * (S_DIM / 32);   // 16384 waves
  const int blocks = tiles / 8;                      // 8 waves per block

  // P0: folded matrices
  gt_kernel<<<S_DIM, S_DIM, 0, stream>>>(Km, Hm, Gt);
  mt_kernel<<<S_DIM, S_DIM, 0, stream>>>(Gt, Fm, Mt);
  // P1: Bu = inputs @ B^T
  gemm_wt_bf16<<<blocks, 256, 0, stream>>>(inp, Bm, nullptr, Bu, M_ROWS, S_DIM, S_DIM);
  // P2: Ko = obs @ K^T ; c = Bu @ Gt^T + Ko   (in place in Cbuf)
  gemm_wt_bf16<<<blocks, 256, 0, stream>>>(obs, Km, nullptr, Cbuf, M_ROWS, S_DIM, S_DIM);
  gemm_wt_bf16<<<blocks, 256, 0, stream>>>(Bu, Gt, Cbuf, Cbuf, M_ROWS, S_DIM, S_DIM);
  // P3: sequential recurrence
  init_copy<<<BATCH_N, S_DIM, 0, stream>>>(init, Sprev);
  kf_recurrence<<<1, 512, 0, stream>>>(init, Cbuf, Mt, out_states, Sprev);
  // P4: tmp = sprev @ F^T + Bu ; y = tmp @ H^T
  gemm_wt_bf16<<<blocks, 256, 0, stream>>>(Sprev, Fm, Bu, Cbuf, M_ROWS, S_DIM, S_DIM);
  gemm_wt_bf16<<<blocks, 256, 0, stream>>>(Cbuf, Hm, nullptr, out_obs, M_ROWS, S_DIM, S_DIM);
}
